// MetaLearner_24137716204014
// MI455X (gfx1250) — matrix operand order repaired and checked by reading, not compiled, not run on hardware
//
#include <hip/hip_runtime.h>
#include <hip/hip_bf16.h>

typedef __bf16 bf16_t;
typedef bf16_t v16bf __attribute__((ext_vector_type(16)));
typedef float  v8f   __attribute__((ext_vector_type(8)));

#define HID 128

__device__ __forceinline__ float warp_sum(float v) {
#pragma unroll
  for (int m = 16; m >= 1; m >>= 1) v += __shfl_xor(v, m, 32);
  return v;
}

// ---------------------------------------------------------------------------
// Async staging: copy a 128-row x 32-col bf16 slab (8 KB) of W into LDS using
// GLOBAL_LOAD_ASYNC_TO_LDS_B128 (ASYNCcnt-tracked, no VGPR round trip).
// 512 16-byte chunks, 2 per thread (blockDim = 256).
// ---------------------------------------------------------------------------
__device__ __forceinline__ void issue_slab_async(
    const bf16_t* __restrict__ Wbf, int ldw, int k0, unsigned ldsBase, int tid)
{
#pragma unroll
  for (int t = 0; t < 2; ++t) {
    const int c = tid + t * 256;           // chunk id in [0,512)
    const int row = c >> 2;                // 0..127
    const int seg = c & 3;                 // 16B segment within 64B row
    const bf16_t* g = Wbf + (long)row * ldw + k0 + seg * 8;
    const unsigned l = ldsBase + (unsigned)c * 16;
    asm volatile("global_load_async_to_lds_b128 %0, %1, off"
                 :: "v"(l), "v"(g) : "memory");
  }
}

__device__ __forceinline__ void wait_async_all() {
  asm volatile("s_wait_asynccnt 0x0" ::: "memory");
}

// ---------------------------------------------------------------------------
// WMMA fragment layouts (wave32):
// A 16x32 bf16: lane L holds row M=L%16; lane-half (L/16) picks
//   K = base+{0..15} (half 0) or base+{16..31} (half 1), the order used here.
// B 32x16 bf16: lane L holds col N=L%16; lane-half picks K {0..15} / {16..31}.
// D 16x16 f32: VGPR r, lane L -> row r + (L/16)*8, col L%16.
// ---------------------------------------------------------------------------

// Pipelined: A fp32 from global (cvt to bf16), W bf16 staged through 2 LDS slabs.
__device__ __forceinline__ void gemm_f32A_ldsW(
    const float* __restrict__ arow,     // per-lane A row base (fp32, stride 1 in K)
    const bf16_t* __restrict__ Wbf, int K,
    bf16_t* slab0, bf16_t* slab1, int tid, int laneLo, int laneHi, v8f acc[8])
{
  const unsigned base0 = (unsigned)(uintptr_t)slab0;
  const unsigned base1 = (unsigned)(uintptr_t)slab1;
  issue_slab_async(Wbf, K, 0, base0, tid);
  const int nsteps = K / 32;
  for (int i = 0; i < nsteps; ++i) {
    const int k0 = i * 32;
    wait_async_all();
    __syncthreads();                      // slab[i&1] ready; slab[(i+1)&1] free
    if (i + 1 < nsteps)
      issue_slab_async(Wbf, K, k0 + 32, (i & 1) ? base0 : base1, tid);

    const int ka = k0 + laneHi * 16;
    v16bf afrag;
#pragma unroll
    for (int j = 0; j < 8; ++j) {
      afrag[j]     = (bf16_t)arow[ka + j];
      afrag[8 + j] = (bf16_t)arow[ka + 8 + j];
    }
    const bf16_t* sl = (i & 1) ? slab1 : slab0;
#pragma unroll
    for (int ct = 0; ct < 8; ++ct) {
      const bf16_t* brow = sl + (ct * 16 + laneLo) * 32 + laneHi * 16;
      v16bf bfrag;
#pragma unroll
      for (int j = 0; j < 16; ++j) bfrag[j] = brow[j];
      acc[ct] = __builtin_amdgcn_wmma_f32_16x16x32_bf16(
          false, afrag, false, bfrag, (short)0, acc[ct], false, false);
    }
  }
  __syncthreads();                        // slabs reusable by caller afterwards
}

__device__ __forceinline__ void store_tile(
    float* __restrict__ Out, long rowBase, int laneLo, int laneHi,
    const float* __restrict__ bias0, const float* __restrict__ bias1, v8f acc[8])
{
#pragma unroll
  for (int ct = 0; ct < 8; ++ct) {
    const int col = ct * 16 + laneLo;
    float bvv = 0.0f;
    if (bias0) bvv += bias0[col];
    if (bias1) bvv += bias1[col];
#pragma unroll
    for (int r = 0; r < 8; ++r) {
      const long row = rowBase + r + laneHi * 8;
      Out[row * HID + col] = acc[ct][r] + bvv;
    }
  }
}

// ---------------------------------------------------------------------------
// fp32 -> bf16 conversion (weights, once per launch)
// ---------------------------------------------------------------------------
__global__ void f32_to_bf16_kernel(const float* __restrict__ X,
                                   bf16_t* __restrict__ Y, int n)
{
  const int i = blockIdx.x * blockDim.x + threadIdx.x;
  if (i < n) Y[i] = (bf16_t)X[i];
}

// ---------------------------------------------------------------------------
// Kernel 1: c = img @ Wi^T + txt @ Wt^T + (bi + bt) for all 17408 tokens.
// Rows [0,1024): support tokens; rows [1024,17408): query tokens.
// ---------------------------------------------------------------------------
__global__ void proj_c_kernel(
    const float* __restrict__ simg, const float* __restrict__ stxt,
    const float* __restrict__ qimg, const float* __restrict__ qtxt,
    const bf16_t* __restrict__ WiB, const float* __restrict__ bi,
    const bf16_t* __restrict__ WtB, const float* __restrict__ bt,
    float* __restrict__ C)
{
  __shared__ bf16_t slab[2][128 * 32];    // 2 x 8KB double buffer
  const int tid = threadIdx.x;
  const int lane = tid & 31;
  const int wave = tid >> 5;
  const int laneLo = lane & 15, laneHi = lane >> 4;
  const long rowBase = (long)blockIdx.x * 128 + wave * 16;
  long aRow = rowBase + laneLo;
  const float* Ai;
  const float* At;
  if (rowBase < 1024) { Ai = simg; At = stxt; }
  else { Ai = qimg; At = qtxt; aRow -= 1024; }

  v8f acc[8];
#pragma unroll
  for (int ct = 0; ct < 8; ++ct) acc[ct] = (v8f){0,0,0,0,0,0,0,0};

  gemm_f32A_ldsW(Ai + aRow * 2048L, WiB, 2048, slab[0], slab[1], tid, laneLo, laneHi, acc);
  gemm_f32A_ldsW(At + aRow * 768L,  WtB,  768, slab[0], slab[1], tid, laneLo, laneHi, acc);
  store_tile(C, rowBase, laneLo, laneHi, bi, bt, acc);
}

// ---------------------------------------------------------------------------
// Generic Out(Mx128 f32) = A(Mx128 bf16) @ W(128x128 bf16)^T + bias.
// Whole W (32 KB) staged into LDS once per block via async loads.
// ---------------------------------------------------------------------------
__global__ void gemm128_bf16_kernel(
    const bf16_t* __restrict__ A, const bf16_t* __restrict__ WB,
    const float* __restrict__ bias, float* __restrict__ Out)
{
  __shared__ bf16_t wsm[HID * HID];       // 32 KB
  const int tid = threadIdx.x;
  const unsigned base = (unsigned)(uintptr_t)wsm;
#pragma unroll
  for (int t = 0; t < 8; ++t) {
    const int c = tid + t * 256;          // 2048 chunks of 16B
    const bf16_t* g = WB + c * 8;
    const unsigned l = base + (unsigned)c * 16;
    asm volatile("global_load_async_to_lds_b128 %0, %1, off"
                 :: "v"(l), "v"(g) : "memory");
  }
  wait_async_all();
  __syncthreads();

  const int lane = tid & 31;
  const int wave = tid >> 5;
  const int laneLo = lane & 15, laneHi = lane >> 4;
  const long rowBase = (long)blockIdx.x * 128 + wave * 16;
  const bf16_t* arow = A + (rowBase + laneLo) * (long)HID;

  v8f acc[8];
#pragma unroll
  for (int ct = 0; ct < 8; ++ct) acc[ct] = (v8f){0,0,0,0,0,0,0,0};

#pragma unroll
  for (int k0 = 0; k0 < HID; k0 += 32) {
    const int ka = k0 + laneHi * 16;
    v16bf afrag;
#pragma unroll
    for (int j = 0; j < 8; ++j) {
      afrag[j]     = arow[ka + j];
      afrag[8 + j] = arow[ka + 8 + j];
    }
    const int kb = k0 + laneHi * 16;
#pragma unroll
    for (int ct = 0; ct < 8; ++ct) {
      const bf16_t* brow = wsm + (ct * 16 + laneLo) * HID + kb;
      v16bf bfrag;
#pragma unroll
      for (int j = 0; j < 16; ++j) bfrag[j] = brow[j];
      acc[ct] = __builtin_amdgcn_wmma_f32_16x16x32_bf16(
          false, afrag, false, bfrag, (short)0, acc[ct], false, false);
    }
  }
  store_tile(Out, rowBase, laneLo, laneHi, bias, nullptr, acc);
}

// ---------------------------------------------------------------------------
// LayerNorm over rows of 128 (fp32 in -> bf16 out). One wave per row.
// ---------------------------------------------------------------------------
__global__ void ln_rows_kernel(
    const float* __restrict__ X, const float* __restrict__ g,
    const float* __restrict__ b, bf16_t* __restrict__ Y, int nrows)
{
  const int lane = threadIdx.x & 31;
  const int row = blockIdx.x * (blockDim.x >> 5) + (threadIdx.x >> 5);
  if (row >= nrows) return;
  const float* x = X + (long)row * HID;
  float v[4];
#pragma unroll
  for (int j = 0; j < 4; ++j) v[j] = x[lane * 4 + j];
  const float mean = warp_sum(v[0] + v[1] + v[2] + v[3]) * (1.0f / 128.0f);
  float q = 0.0f;
#pragma unroll
  for (int j = 0; j < 4; ++j) { const float d = v[j] - mean; q += d * d; }
  const float var = warp_sum(q) * (1.0f / 128.0f);
  const float inv = rsqrtf(var + 1e-5f);
  bf16_t* y = Y + (long)row * HID;
#pragma unroll
  for (int j = 0; j < 4; ++j) {
    const int c = lane * 4 + j;
    y[c] = (bf16_t)((v[j] - mean) * inv * g[c] + b[c]);
  }
}

// ---------------------------------------------------------------------------
// L2-normalize rows of 128 (fp32 -> fp32). One wave per row.
// ---------------------------------------------------------------------------
__global__ void l2norm_rows_kernel(
    const float* __restrict__ X, float* __restrict__ Y, int nrows)
{
  const int lane = threadIdx.x & 31;
  const int row = blockIdx.x * (blockDim.x >> 5) + (threadIdx.x >> 5);
  if (row >= nrows) return;
  const float* x = X + (long)row * HID;
  float v[4];
  float ss = 0.0f;
#pragma unroll
  for (int j = 0; j < 4; ++j) { v[j] = x[lane * 4 + j]; ss += v[j] * v[j]; }
  const float nrm = sqrtf(warp_sum(ss));
  const float inv = 1.0f / fmaxf(nrm, 1e-8f);
  float* y = Y + (long)row * HID;
#pragma unroll
  for (int j = 0; j < 4; ++j) y[lane * 4 + j] = v[j] * inv;
}

// ---------------------------------------------------------------------------
// Support attention: per task, 4x4 attention over HID=128, then LN2 -> bf16.
// One block (128 threads) per task.
// ---------------------------------------------------------------------------
__device__ __forceinline__ float block_sum128(float v, float* scratch, int tid) {
  const float w = warp_sum(v);
  if ((tid & 31) == 0) scratch[tid >> 5] = w;
  __syncthreads();
  const float total = scratch[0] + scratch[1] + scratch[2] + scratch[3];
  __syncthreads();
  return total;
}

__global__ void support_attn_kernel(
    const float* __restrict__ Q, const float* __restrict__ Kk,
    const float* __restrict__ V, const float* __restrict__ g2,
    const float* __restrict__ b2, bf16_t* __restrict__ CtxLn)
{
  const int t = blockIdx.x;
  const int tid = threadIdx.x;
  const float* q = Q  + (long)t * 4 * HID;
  const float* k = Kk + (long)t * 4 * HID;
  const float* v = V  + (long)t * 4 * HID;
  __shared__ float sa[4][4];
  __shared__ float red4[4];
  const float inv_scale = 1.0f / (sqrtf(128.0f) + 1e-8f);

  if (tid < 16) {
    const int i = tid >> 2, j = tid & 3;
    float d = 0.0f;
    for (int x = 0; x < HID; ++x) d += q[i * HID + x] * k[j * HID + x];
    sa[i][j] = d * inv_scale;
  }
  __syncthreads();
  if (tid < 4) {
    float m = fmaxf(fmaxf(sa[tid][0], sa[tid][1]), fmaxf(sa[tid][2], sa[tid][3]));
    float e[4], se = 0.0f;
#pragma unroll
    for (int j = 0; j < 4; ++j) { e[j] = expf(sa[tid][j] - m); se += e[j]; }
    float a[4], s2 = 0.0f;
#pragma unroll
    for (int j = 0; j < 4; ++j) { a[j] = e[j] / se + 1e-10f; s2 += a[j]; }
#pragma unroll
    for (int j = 0; j < 4; ++j) {
      float av = a[j] / s2;
      av = fminf(fmaxf(av, 1e-7f), 1.0f);
      sa[tid][j] = av;
    }
  }
  __syncthreads();
  float ctx[4];
#pragma unroll
  for (int i = 0; i < 4; ++i)
    ctx[i] = sa[i][0] * v[0 * HID + tid] + sa[i][1] * v[1 * HID + tid]
           + sa[i][2] * v[2 * HID + tid] + sa[i][3] * v[3 * HID + tid];

  const float gg = g2[tid], bb = b2[tid];
#pragma unroll
  for (int i = 0; i < 4; ++i) {
    const float mean = block_sum128(ctx[i], red4, tid) * (1.0f / 128.0f);
    const float d = ctx[i] - mean;
    const float var = block_sum128(d * d, red4, tid) * (1.0f / 128.0f);
    const float y = d * rsqrtf(var + 1e-5f) * gg + bb;
    CtxLn[((long)t * 4 + i) * HID + tid] = (bf16_t)y;
  }
}

// ---------------------------------------------------------------------------
// Logits: per task, qn(64x128) . pn(4x128)^T * 10. One block (256 thr) per task.
// ---------------------------------------------------------------------------
__global__ void logits_kernel(const float* __restrict__ Fn, float* __restrict__ Out)
{
  const int t = blockIdx.x;
  const int tid = threadIdx.x;           // tid = qi*4 + c
  const int qi = tid >> 2, c = tid & 3;
  const float* qn = Fn + (1024 + (long)t * 64 + qi) * HID;
  const float* pn = Fn + ((long)t * 4 + c) * HID;
  float d = 0.0f;
#pragma unroll 8
  for (int x = 0; x < HID; ++x) d += qn[x] * pn[x];
  Out[(long)t * 256 + tid] = d * 10.0f;
}

// ---------------------------------------------------------------------------
extern "C" void kernel_launch(void* const* d_in, const int* in_sizes, int n_in,
                              void* d_out, int out_size, void* d_ws, size_t ws_size,
                              hipStream_t stream)
{
  const float* simg = (const float*)d_in[0];
  const float* stxt = (const float*)d_in[1];
  // d_in[2] = support_labels (unused by reference)
  const float* qimg = (const float*)d_in[3];
  const float* qtxt = (const float*)d_in[4];
  const float* Wi = (const float*)d_in[5];
  const float* bi = (const float*)d_in[6];
  const float* Wt = (const float*)d_in[7];
  const float* bt = (const float*)d_in[8];
  const float* g1 = (const float*)d_in[9];
  const float* b1 = (const float*)d_in[10];
  const float* Wq = (const float*)d_in[11];
  const float* bq = (const float*)d_in[12];
  const float* Wk = (const float*)d_in[13];
  const float* bk = (const float*)d_in[14];
  const float* Wv = (const float*)d_in[15];
  const float* bv = (const float*)d_in[16];
  const float* g2 = (const float*)d_in[17];
  const float* b2 = (const float*)d_in[18];
  const float* Wo = (const float*)d_in[19];
  const float* bo = (const float*)d_in[20];
  float* out = (float*)d_out;
  (void)in_sizes; (void)n_in; (void)out_size; (void)ws_size;

  const long Msup = 1024, Mq = 16384, Mtot = Msup + Mq; // 17408 tokens

  char* ws = (char*)d_ws;
  size_t off = 0;
  auto carve = [&](size_t bytes) -> void* {
    void* p = ws + off;
    off = (off + bytes + 255) & ~(size_t)255;
    return p;
  };
  float*  cbuf   = (float*) carve((size_t)Mtot * HID * 4);
  bf16_t* cln    = (bf16_t*)carve((size_t)Mtot * HID * 2);
  float*  qbuf   = (float*) carve((size_t)Msup * HID * 4);
  float*  kbuf   = (float*) carve((size_t)Msup * HID * 4);
  float*  vbuf   = (float*) carve((size_t)Mtot * HID * 4);
  bf16_t* ctxln  = (bf16_t*)carve((size_t)Mtot * HID * 2);
  float*  feats  = (float*) carve((size_t)Mtot * HID * 4);
  float*  featsn = (float*) carve((size_t)Mtot * HID * 4);
  bf16_t* WiB    = (bf16_t*)carve((size_t)HID * 2048 * 2);
  bf16_t* WtB    = (bf16_t*)carve((size_t)HID * 768 * 2);
  bf16_t* WqB    = (bf16_t*)carve((size_t)HID * HID * 2);
  bf16_t* WkB    = (bf16_t*)carve((size_t)HID * HID * 2);
  bf16_t* WvB    = (bf16_t*)carve((size_t)HID * HID * 2);
  bf16_t* WoB    = (bf16_t*)carve((size_t)HID * HID * 2);

  dim3 b256(256), b128(128);

  // 0) one-time weight conversion to bf16
  f32_to_bf16_kernel<<<(HID * 2048 + 255) / 256, b256, 0, stream>>>(Wi, WiB, HID * 2048);
  f32_to_bf16_kernel<<<(HID * 768  + 255) / 256, b256, 0, stream>>>(Wt, WtB, HID * 768);
  f32_to_bf16_kernel<<<(HID * HID  + 255) / 256, b256, 0, stream>>>(Wq, WqB, HID * HID);
  f32_to_bf16_kernel<<<(HID * HID  + 255) / 256, b256, 0, stream>>>(Wk, WkB, HID * HID);
  f32_to_bf16_kernel<<<(HID * HID  + 255) / 256, b256, 0, stream>>>(Wv, WvB, HID * HID);
  f32_to_bf16_kernel<<<(HID * HID  + 255) / 256, b256, 0, stream>>>(Wo, WoB, HID * HID);

  // 1) c = img@Wi^T + txt@Wt^T + bi + bt (all tokens)  [dominant, memory-bound]
  proj_c_kernel<<<(int)(Mtot / 128), b256, 0, stream>>>(
      simg, stxt, qimg, qtxt, WiB, bi, WtB, bt, cbuf);
  // 2) LN1 -> bf16
  ln_rows_kernel<<<(int)((Mtot + 7) / 8), b256, 0, stream>>>(cbuf, g1, b1, cln, (int)Mtot);
  // 3) q,k (support only), v (all tokens)
  gemm128_bf16_kernel<<<(int)(Msup / 128), b256, 0, stream>>>(cln, WqB, bq, qbuf);
  gemm128_bf16_kernel<<<(int)(Msup / 128), b256, 0, stream>>>(cln, WkB, bk, kbuf);
  gemm128_bf16_kernel<<<(int)(Mtot / 128), b256, 0, stream>>>(cln, WvB, bv, vbuf);
  // 4) support attention + LN2 -> ctxln rows [0,1024)
  support_attn_kernel<<<256, b128, 0, stream>>>(qbuf, kbuf, vbuf, g2, b2, ctxln);
  // 5) query ctx == v (length-1 softmax is exactly 1): LN2 of v rows
  ln_rows_kernel<<<(int)((Mq + 7) / 8), b256, 0, stream>>>(
      vbuf + Msup * HID, g2, b2, ctxln + Msup * HID, (int)Mq);
  // 6) feats = ctxln @ Wo^T + bo (all tokens)
  gemm128_bf16_kernel<<<(int)(Mtot / 128), b256, 0, stream>>>(ctxln, WoB, bo, feats);
  // 7) L2 normalize rows
  l2norm_rows_kernel<<<(int)((Mtot + 7) / 8), b256, 0, stream>>>(feats, featsn, (int)Mtot);
  // 8) logits = qn . pn^T * 10
  logits_kernel<<<256, b256, 0, stream>>>(featsn, out);
}
